// RetLayer_25503515804191
// MI455X (gfx1250) — matrix operand order repaired and checked by reading, not compiled, not run on hardware
//
#include <hip/hip_runtime.h>
#include <hip/hip_bf16.h>
#include <math.h>

// ---------------- problem constants (match setup_inputs) ----------------
#define B_    16
#define S_    1056
#define DM_   256
#define H_    8
#define DK_   32
#define TXT_  32
#define VID_  1024
#define MT_   (B_ * S_)          // 16896 rows
#define QTILES_ (S_ / 16)        // 66
#define QSCALE 0.17677669529663687f   // 1/sqrt(32)
#define LOG2_GAMMA (-0.15200309344504995f) // log2(0.9)
#define LOG2_1E4   (13.287712379549449f)   // log2(10000)

typedef __attribute__((ext_vector_type(16))) __bf16 bf16x16;
typedef __attribute__((ext_vector_type(8)))  float  f32x8;

union BfVec { uint4 u[2]; bf16x16 v; };

// A tile (16x32 bf16, row-major, row ptr at column k0): lane = (m, half)
// elements 0..7  = K[8*half .. 8*half+7], elements 8..15 = K[16+8*half .. +7]
__device__ __forceinline__ bf16x16 load_a32(const __bf16* __restrict__ rowk0, int half) {
    BfVec t;
    t.u[0] = *reinterpret_cast<const uint4*>(rowk0 + 8 * half);
    t.u[1] = *reinterpret_cast<const uint4*>(rowk0 + 16 + 8 * half);
    return t.v;
}
// B tile (32x16 bf16) fed from Bt (N x K row-major, row ptr at column k0):
// lane = (n, half), elements 0..7 = K[8*half .. 8*half+7], elements 8..15 = K[16+8*half .. +7]
__device__ __forceinline__ bf16x16 load_b32(const __bf16* __restrict__ rowk0, int half) {
    BfVec t;
    t.u[0] = *reinterpret_cast<const uint4*>(rowk0 + 8 * half);
    t.u[1] = *reinterpret_cast<const uint4*>(rowk0 + 16 + 8 * half);
    return t.v;
}

// gfx1250 async copy: 16B global -> LDS, tracked on ASYNCcnt.
// Low 32 bits of a flat pointer into LDS are the LDS byte offset
// (shared aperture base is 4GB-aligned per CDNA5 ISA 10.2).
__device__ __forceinline__ void async_g2l_b128(const void* gptr, void* lptr) {
    unsigned lds_off = (unsigned)(unsigned long long)lptr;
    unsigned long long ga = (unsigned long long)gptr;
    asm volatile("global_load_async_to_lds_b128 %0, %1, off"
                 :: "v"(lds_off), "v"(ga) : "memory");
}
__device__ __forceinline__ void wait_asynccnt0() {
    asm volatile("s_wait_asynccnt 0x0" ::: "memory");
}

// ---------------- kernel 0: zero the loss scalar ----------------
__global__ void k_zero_loss(float* loss) { *loss = 0.0f; }

// ---------------- kernel 1: retention mask + sparsity loss ----------------
__global__ void k_mask(const float* __restrict__ lvr,
                       const int* __restrict__ ridx,
                       const int* __restrict__ cidx,
                       float* __restrict__ mask,
                       float* __restrict__ loss) {
    int idx = blockIdx.x * 256 + threadIdx.x;
    float av = 0.0f;
    if (idx < S_ * S_) {
        int n = idx / S_, m = idx - n * S_;
        float t  = lvr[(size_t)ridx[idx] * VID_ + cidx[idx]];
        float lr = 1.0f / (1.0f + __expf(-t));
        float vret = (n == m) ? 0.0f : lr;
        float dmat = (n >= m) ? exp2f((float)(n - m) * LOG2_GAMMA) : 0.0f;
        float lm   = (n == m) ? 1.0f : vret;
        mask[idx] = dmat * lm;
        av = fabsf(vret);
    }
    __shared__ float red[256];
    red[threadIdx.x] = av;
    __syncthreads();
    for (int s = 128; s > 0; s >>= 1) {
        if (threadIdx.x < s) red[threadIdx.x] += red[threadIdx.x + s];
        __syncthreads();
    }
    if (threadIdx.x == 0)
        atomicAdd(loss, red[0] * (1.0f / ((float)S_ * (float)S_)));
}

// ---------------- kernel 2: positional embeds -> bf16 GEMM inputs ----------------
__global__ void k_prep_inputs(const float* __restrict__ src,
                              __bf16* __restrict__ AQ,
                              __bf16* __restrict__ AK,
                              __bf16* __restrict__ AV) {
    size_t idx = (size_t)blockIdx.x * 256 + threadIdx.x; // over B*S*128 (pairs)
    if (idx >= (size_t)B_ * S_ * 128) return;
    int i = (int)(idx & 127);
    int s = (int)((idx >> 7) % S_);
    int b = (int)(idx / ((size_t)128 * S_));
    size_t base = ((size_t)b * S_ + s) * DM_ + 2 * i;
    float x0 = src[base], x1 = src[base + 1];
    AV[base]     = (__bf16)x0;
    AV[base + 1] = (__bf16)x1;
    float inv_freq = exp2f(-((float)i * (1.0f / 128.0f)) * LOG2_1E4); // 10000^(-i/128)
    if (s < TXT_) {
        float xe = (float)(s + 1) * (6.2831853071795864f / (32.0f + 1e-6f));
        float sn, cs;
        __sincosf(xe * inv_freq, &sn, &cs);
        float v0 = sn + x0, v1 = cs + x1;
        AQ[base] = (__bf16)v0; AQ[base + 1] = (__bf16)v1;
        AK[base] = (__bf16)v0; AK[base + 1] = (__bf16)v1;
    } else {
        int j = s - TXT_;
        float basef = ((float)(2 * i) + 0.4f * 256.0f) / (1.4f * 256.0f);
        float scq = exp2f(((float)j * (1.0f / 512.0f)) * log2f(basef));
        float sck = 1.0f / scq;
        float sn, cs;
        __sincosf((float)j * inv_freq, &sn, &cs);
        float q0 = (x0 * cs - x1 * sn) * scq;
        float q1 = (x1 * cs + x0 * sn) * scq;
        float k0 = (x0 * cs - x1 * sn) * sck;
        float k1 = (x1 * cs + x0 * sn) * sck;
        AQ[base] = (__bf16)q0; AQ[base + 1] = (__bf16)q1;
        AK[base] = (__bf16)k0; AK[base + 1] = (__bf16)k1;
    }
}

// ---------------- kernel 3: weights -> bf16 (optionally transposed N x K) ----------------
__global__ void k_prep_w(const float* __restrict__ W, __bf16* __restrict__ Wt, int transpose) {
    int idx = blockIdx.x * 256 + threadIdx.x;
    if (idx >= DM_ * DM_) return;
    int n = idx / DM_, k = idx - n * DM_;
    Wt[idx] = (__bf16)(transpose ? W[k * DM_ + n] : W[idx]);
}

// ---------------- kernel 4: WMMA projection GEMM (Q/K/V) ----------------
// mode 0/1: out[(b*H+h)*S + s][d]   (per-head row-major, Q gets 1/sqrt(dk))
// mode 2  : out[(b*H+h)*DK + d][s]  (per-head TRANSPOSED -> B-operand for PV)
__global__ void k_proj(const __bf16* __restrict__ A,
                       const __bf16* __restrict__ Wt_txt,
                       const __bf16* __restrict__ Wt_vid,
                       __bf16* __restrict__ outp, int mode, float scale) {
    int n0 = blockIdx.x * 16;
    int m0 = blockIdx.y * 16;
    int lane = threadIdx.x;
    int half = lane >> 4, mr = lane & 15;
    int b  = m0 / S_;
    int s0 = m0 - b * S_;
    const __bf16* Wt = (s0 < TXT_) ? Wt_txt : Wt_vid;
    const __bf16* arow = A  + (size_t)(m0 + mr) * DM_;
    const __bf16* brow = Wt + (size_t)(n0 + mr) * DM_;
    f32x8 c = {};
#pragma unroll
    for (int k0 = 0; k0 < DM_; k0 += 32) {
        bf16x16 av = load_a32(arow + k0, half);
        bf16x16 bv = load_b32(brow + k0, half);
        c = __builtin_amdgcn_wmma_f32_16x16x32_bf16(false, av, false, bv,
                                                    (short)0, c, false, false);
    }
#pragma unroll
    for (int r = 0; r < 8; ++r) {
        int M  = 8 * half + r;
        int gm = m0 + M;
        int bb = gm / S_;
        int s  = gm - bb * S_;
        int col = n0 + mr;
        int h = col >> 5, d = col & 31;
        float v = c[r] * scale;
        if (mode == 2)
            outp[(((size_t)bb * H_ + h) * DK_ + d) * S_ + s] = (__bf16)v;
        else
            outp[(((size_t)bb * H_ + h) * S_ + s) * DK_ + d] = (__bf16)v;
    }
}

// ---------------- kernel 5: fused WMMA attention ----------------
// One block per (batch, 16-row q-tile); 16 waves; all 8 heads in-block.
// Mask row-block async-copied to LDS once (8x reuse); att accumulated in
// LDS across heads and written to HBM exactly once per element.
__global__ __launch_bounds__(512)
void k_attn(const __bf16* __restrict__ Qh, const __bf16* __restrict__ Kh,
            const __bf16* __restrict__ Vt, const float* __restrict__ maskp,
            __bf16* __restrict__ ctxA, float* __restrict__ attOut) {
    __shared__ float  sc[16][S_];        // masked scores -> exp values
    __shared__ float  lmask[16][S_];     // mask tile (async-staged, 8x reuse)
    __shared__ float  attacc[16][S_];    // head-summed probabilities
    __shared__ __bf16 pbuf[16][S_];      // normalized probs (bf16 A-operand)
    __shared__ float  ctxs[16][DK_];     // PV partial-sum reduction
    int b  = blockIdx.x / QTILES_;
    int qt = blockIdx.x - b * QTILES_;
    int q0 = qt * 16;
    int tid  = threadIdx.x;
    int wave = tid >> 5, lane = tid & 31;
    int half = lane >> 4, mr = lane & 15;

    // ---- stage mask rows into LDS with gfx1250 async b128 copies ----
    {
        const int CPR = S_ / 4;               // 264 16-byte chunks per row
        for (int i = tid; i < 16 * CPR; i += 512) {
            int r = i / CPR, c = i - r * CPR;
            async_g2l_b128(maskp + (size_t)(q0 + r) * S_ + c * 4, &lmask[r][c * 4]);
        }
    }
    for (int i = tid; i < 16 * S_; i += 512) ((float*)attacc)[i] = 0.0f;
    wait_asynccnt0();
    __syncthreads();

    for (int h = 0; h < H_; ++h) {
        int bh = b * H_ + h;
        // ---- scores: q(16x32) x k^T, tiles split across 16 waves ----
        const __bf16* qrow = Qh + ((size_t)bh * S_ + q0 + mr) * DK_;
        bf16x16 aq = load_a32(qrow, half);
        for (int kt = wave; kt < QTILES_; kt += 16) {
            int kc = kt * 16;
            const __bf16* krow = Kh + ((size_t)bh * S_ + kc + mr) * DK_;
            bf16x16 bv = load_b32(krow, half);
            f32x8 c = {};
            c = __builtin_amdgcn_wmma_f32_16x16x32_bf16(false, aq, false, bv,
                                                        (short)0, c, false, false);
#pragma unroll
            for (int r = 0; r < 8; ++r) {
                int M = 8 * half + r;
                int col = kc + mr;
                sc[M][col] = c[r] * lmask[M][col];
            }
        }
        __syncthreads();

        // ---- softmax: one wave per row; accumulate att in LDS ----
        {
            int row = wave;
            float mx = -3.4e38f;
            for (int k = lane; k < S_; k += 32) mx = fmaxf(mx, sc[row][k]);
            for (int off = 16; off >= 1; off >>= 1) mx = fmaxf(mx, __shfl_xor(mx, off, 32));
            float sum = 0.0f;
            for (int k = lane; k < S_; k += 32) {
                float e = __expf(sc[row][k] - mx);
                sc[row][k] = e;
                sum += e;
            }
            for (int off = 16; off >= 1; off >>= 1) sum += __shfl_xor(sum, off, 32);
            float inv = 1.0f / sum;
            for (int k = lane; k < S_; k += 32) {
                float p = sc[row][k] * inv;
                pbuf[row][k] = (__bf16)p;
                attacc[row][k] += p * 0.125f;
            }
            ((float*)ctxs)[tid] = 0.0f;   // 16*32 floats == 512 threads
        }
        __syncthreads();

        // ---- PV: probs(16x1056) x v(1056x32); K-chunks split across waves ----
        {
            int nt = wave & 1;            // which 16-wide slice of dk
            f32x8 c = {};
            for (int ck = (wave >> 1); ck < 33; ck += 8) {
                int kc = ck * 32;
                BfVec ta;
                ta.u[0] = *reinterpret_cast<const uint4*>(&pbuf[mr][kc + 8 * half]);
                ta.u[1] = *reinterpret_cast<const uint4*>(&pbuf[mr][kc + 16 + 8 * half]);
                const __bf16* vrow = Vt + ((size_t)bh * DK_ + nt * 16 + mr) * S_ + kc;
                bf16x16 bv = load_b32(vrow, half);
                c = __builtin_amdgcn_wmma_f32_16x16x32_bf16(false, ta.v, false, bv,
                                                            (short)0, c, false, false);
            }
#pragma unroll
            for (int r = 0; r < 8; ++r) {
                int M = 8 * half + r;
                atomicAdd(&ctxs[M][nt * 16 + mr], c[r]);   // ds_add_f32
            }
        }
        __syncthreads();

        // ---- emit ctx tile (bf16, packed [b][s][h*32+d]) ----
        {
            int M = tid >> 5, d = tid & 31;
            ctxA[((size_t)b * S_ + q0 + M) * DM_ + h * DK_ + d] = (__bf16)ctxs[M][d];
        }
        __syncthreads();
    }

    // ---- single streaming write of the head-averaged attention ----
    for (int i = tid; i < 16 * S_; i += 512) {
        int r = i / S_, k = i - r * S_;
        attOut[((size_t)b * S_ + q0 + r) * S_ + k] = attacc[r][k];
    }
}

// ---------------- kernel 6: WMMA output projection (+bias) ----------------
__global__ void k_out(const __bf16* __restrict__ A, const __bf16* __restrict__ Wo,
                      const float* __restrict__ bias, float* __restrict__ out) {
    int n0 = blockIdx.x * 16;
    int m0 = blockIdx.y * 16;
    int lane = threadIdx.x;
    int half = lane >> 4, mr = lane & 15;
    const __bf16* arow = A  + (size_t)(m0 + mr) * DM_;
    const __bf16* brow = Wo + (size_t)(n0 + mr) * DM_;   // W_out is already N x K
    f32x8 c = {};
#pragma unroll
    for (int k0 = 0; k0 < DM_; k0 += 32) {
        bf16x16 av = load_a32(arow + k0, half);
        bf16x16 bv = load_b32(brow + k0, half);
        c = __builtin_amdgcn_wmma_f32_16x16x32_bf16(false, av, false, bv,
                                                    (short)0, c, false, false);
    }
#pragma unroll
    for (int r = 0; r < 8; ++r) {
        int M = 8 * half + r;
        int col = n0 + mr;
        out[(size_t)(m0 + M) * DM_ + col] = c[r] + bias[col];
    }
}

// ---------------- host-side launcher ----------------
extern "C" void kernel_launch(void* const* d_in, const int* in_sizes, int n_in,
                              void* d_out, int out_size, void* d_ws, size_t ws_size,
                              hipStream_t stream) {
    (void)in_sizes; (void)n_in; (void)out_size; (void)ws_size;
    const float* src   = (const float*)d_in[0];
    const float* WtxtQ = (const float*)d_in[1];
    const float* WtxtK = (const float*)d_in[2];
    const float* WtxtV = (const float*)d_in[3];
    const float* WvidQ = (const float*)d_in[4];
    const float* WvidK = (const float*)d_in[5];
    const float* WvidV = (const float*)d_in[6];
    const float* lvr   = (const float*)d_in[7];
    const float* Wout  = (const float*)d_in[8];
    const float* bout  = (const float*)d_in[9];
    const int*   ridx  = (const int*)d_in[10];
    const int*   cidx  = (const int*)d_in[11];

    char* ws = (char*)d_ws;
    size_t off = 0;
    float*  mask = (float*)(ws + off);  off += (size_t)S_ * S_ * sizeof(float);
    __bf16* AQ   = (__bf16*)(ws + off); off += (size_t)MT_ * DM_ * 2;
    __bf16* AK   = (__bf16*)(ws + off); off += (size_t)MT_ * DM_ * 2;
    __bf16* AV   = (__bf16*)(ws + off); off += (size_t)MT_ * DM_ * 2;
    __bf16* Wb[7];
    for (int i = 0; i < 7; ++i) { Wb[i] = (__bf16*)(ws + off); off += (size_t)DM_ * DM_ * 2; }
    __bf16* Qh   = (__bf16*)(ws + off); off += (size_t)B_ * H_ * S_ * DK_ * 2;
    __bf16* Khb  = (__bf16*)(ws + off); off += (size_t)B_ * H_ * S_ * DK_ * 2;
    __bf16* Vt   = (__bf16*)(ws + off); off += (size_t)B_ * H_ * DK_ * S_ * 2;
    __bf16* ctxA = (__bf16*)(ws + off); off += (size_t)MT_ * DM_ * 2;

    float* out  = (float*)d_out;
    float* att  = out + (size_t)MT_ * DM_;
    float* loss = att + (size_t)B_ * S_ * S_;

    k_zero_loss<<<1, 1, 0, stream>>>(loss);
    k_mask<<<(S_ * S_ + 255) / 256, 256, 0, stream>>>(lvr, ridx, cidx, mask, loss);
    k_prep_inputs<<<(B_ * S_ * 128 + 255) / 256, 256, 0, stream>>>(src, AQ, AK, AV);

    // weights -> bf16: 0..5 transposed (N x K) for Q/K/V txt+vid; 6 = W_out as-is
    k_prep_w<<<256, 256, 0, stream>>>(WtxtQ, Wb[0], 1);
    k_prep_w<<<256, 256, 0, stream>>>(WvidQ, Wb[1], 1);
    k_prep_w<<<256, 256, 0, stream>>>(WtxtK, Wb[2], 1);
    k_prep_w<<<256, 256, 0, stream>>>(WvidK, Wb[3], 1);
    k_prep_w<<<256, 256, 0, stream>>>(WtxtV, Wb[4], 1);
    k_prep_w<<<256, 256, 0, stream>>>(WvidV, Wb[5], 1);
    k_prep_w<<<256, 256, 0, stream>>>(Wout,  Wb[6], 0);

    dim3 gg(DM_ / 16, MT_ / 16);
    k_proj<<<gg, 32, 0, stream>>>(AQ, Wb[0], Wb[1], Qh,  0, QSCALE);
    k_proj<<<gg, 32, 0, stream>>>(AK, Wb[2], Wb[3], Khb, 1, 1.0f);
    k_proj<<<gg, 32, 0, stream>>>(AV, Wb[4], Wb[5], Vt,  2, 1.0f);

    k_attn<<<B_ * QTILES_, 512, 0, stream>>>(Qh, Khb, Vt, mask, ctxA, att);

    k_out<<<gg, 32, 0, stream>>>(ctxA, Wb[6], bout, out);
}
